// AttentionLayer_43593918054775
// MI455X (gfx1250) — matrix operand order repaired and checked by reading, not compiled, not run on hardware
//
#include <hip/hip_runtime.h>

// ---------------------------------------------------------------------------
// AttentionLayer for MI455X (gfx1250):
//   * all GEMMs via v_wmma_f32_16x16x32_f16
//   * tile movement via TDM (tensor_load_to_lds) and global_load_async_to_lds
// ---------------------------------------------------------------------------

typedef __attribute__((ext_vector_type(16))) _Float16     v16h;
typedef __attribute__((ext_vector_type(8)))  float        v8f;
typedef __attribute__((ext_vector_type(4)))  unsigned int v4u;
typedef __attribute__((ext_vector_type(8)))  int          v8i;
typedef __attribute__((ext_vector_type(4)))  int          v4i;

#define B_   64
#define TQ_  1024
#define TK_  512
#define H_   512
#define LN10000 9.210340371976184f
#define XSCALE  0.04419417382415922f   // 1/sqrt(512)

__device__ __forceinline__ v8f wmma16x16x32(v16h a, v16h b, v8f c) {
  return __builtin_amdgcn_wmma_f32_16x16x32_f16(false, a, false, b, (short)0, c,
                                                false, false);
}

__device__ __forceinline__ unsigned lds_of(const void* p) {
  // flat LDS addresses carry the LDS byte offset in the low 32 bits
  return (unsigned)(unsigned long long)p;
}

// ---- async global -> LDS (ASYNCcnt path), GVS addressing ----
__device__ __forceinline__ void async_ld_b128(unsigned ldsoff, unsigned voff,
                                              const void* sbase) {
  asm volatile("global_load_async_to_lds_b128 %0, %1, %2"
               :: "v"(ldsoff), "v"(voff), "s"((unsigned long long)sbase)
               : "memory");
}
__device__ __forceinline__ void wait_async0() {
  asm volatile("s_wait_asynccnt 0x0" ::: "memory");
}
__device__ __forceinline__ void wait_ds0() {
  asm volatile("s_wait_dscnt 0x0" ::: "memory");
}

// ---- Tensor Data Mover: 2-D tile load (D# group0 + group1, groups 2/3 zero) --
#if defined(__has_builtin)
#if __has_builtin(__builtin_amdgcn_tensor_load_to_lds)
#define HAVE_TDM 1
#endif
#endif

#ifdef HAVE_TDM
__device__ __forceinline__ void tdm_load_2d(unsigned lds_addr, const void* gaddr,
                                            unsigned tile_d0, unsigned tile_d1,
                                            unsigned tensor_d0, unsigned stride0,
                                            unsigned tensor_d1) {
  unsigned long long ga = (unsigned long long)gaddr;
  v4u g0;
  g0[0] = 1u;                                   // count=1, user-mode descriptor
  g0[1] = lds_addr;                             // lds_addr (bytes)
  g0[2] = (unsigned)(ga & 0xffffffffu);         // global_addr[31:0]
  g0[3] = (unsigned)((ga >> 32) & 0x01ffffffu)  // global_addr[56:32]
          | (2u << 30);                         // type=2 ("image")
  v8i g1;
  g1[0] = (int)(1u << 16);                      // data_size=1 -> 2 bytes/elem
  g1[1] = (int)(tensor_d0 << 16);               // tensor_dim0[15:0]
  g1[2] = (int)((tensor_d0 >> 16) | (tensor_d1 << 16));   // dim0 hi | dim1 lo
  g1[3] = (int)((tensor_d1 >> 16) | (tile_d0 << 16));     // dim1 hi | tile_dim0
  g1[4] = (int)(tile_d1 & 0xffffu);             // tile_dim1 (tile_dim2 = 0)
  g1[5] = (int)stride0;                         // tensor_dim0_stride[31:0]
  g1[6] = 0;                                    // stride0 hi | dim1_stride lo
  g1[7] = 0;
  v4i z = {0, 0, 0, 0};
#if __clang_major__ >= 23
  v8i z8 = {0, 0, 0, 0, 0, 0, 0, 0};
  __builtin_amdgcn_tensor_load_to_lds(g0, g1, z, z, z8, 0);
#else
  __builtin_amdgcn_tensor_load_to_lds(g0, g1, z, z, 0);
#endif
}
__device__ __forceinline__ void wait_tensor0() {
  __builtin_amdgcn_s_wait_tensorcnt(0);
  asm volatile("" ::: "memory");
}
__device__ __forceinline__ void wait_tensor1() {
  __builtin_amdgcn_s_wait_tensorcnt(1);
  asm volatile("" ::: "memory");
}
#else
// fallback: lane-cooperative copy (32 lanes), keeps file compiling without TDM
__device__ __forceinline__ void tdm_load_2d(unsigned lds_addr, const void* gaddr,
                                            unsigned tile_d0, unsigned tile_d1,
                                            unsigned, unsigned stride0, unsigned) {
  _Float16* dst = (_Float16*)(unsigned long long)lds_addr;  // placeholder
  (void)dst; (void)gaddr; (void)tile_d0; (void)tile_d1; (void)stride0;
}
__device__ __forceinline__ void wait_tensor0() {}
__device__ __forceinline__ void wait_tensor1() {}
#endif

// ---------------------------------------------------------------------------
// prep kernels: f32 -> f16 (+ sinusoidal positional encoding)
// ---------------------------------------------------------------------------

__global__ __launch_bounds__(256)
void cvt_f16_kernel(const float* __restrict__ src, _Float16* __restrict__ dst,
                    int n) {
  int i = blockIdx.x * 256 + threadIdx.x;
  if (i < n) dst[i] = (_Float16)src[i];
}

__global__ __launch_bounds__(256)
void prep_pe_kernel(const float* __restrict__ x, const int* __restrict__ pos,
                    _Float16* __restrict__ dst, int T, float w) {
  size_t i = (size_t)blockIdx.x * 256 + threadIdx.x;
  int    c  = (int)(i & (H_ - 1));
  size_t bt = i >> 9;                 // b*T + t
  int    t  = (int)(bt % (size_t)T);
  int    b  = (int)(bt / (size_t)T);
  int    p  = pos[(size_t)b * T + t];
  float pe = 0.0f;
  if (p != 0) {
    float invf = __expf(-(float)(2 * (c >> 1)) * (LN10000 / (float)H_));
    float ang  = (float)p * w * invf;
    pe = (c & 1) ? __cosf(ang) : __sinf(ang);
  }
  dst[i] = (_Float16)(x[i] + pe);
}

// ---------------------------------------------------------------------------
// Tiled f16 GEMM:  C[M x 512] = A[M x 512] @ W[512 x 512] * scale + bias
// 128x128 tile per block, 256 threads (8 waves); LDS staged via async-to-LDS.
// ---------------------------------------------------------------------------

template <bool OUT_F32>
__global__ __launch_bounds__(256)
void gemm_f16_kernel(const _Float16* __restrict__ A,
                     const _Float16* __restrict__ W,
                     const float* __restrict__ bias,
                     float* __restrict__ Cf, _Float16* __restrict__ Ch,
                     float scale) {
  __shared__ _Float16 As[128][32];
  __shared__ _Float16 Ws[32][128];

  const int tid  = threadIdx.x;
  const int lane = tid & 31;
  const int wid  = tid >> 5;
  const int m0   = blockIdx.x * 128;
  const int n0   = blockIdx.y * 128;

  v8f acc[8] = {};

#pragma unroll 1
  for (int kk = 0; kk < H_ / 32; ++kk) {
    // ---- stage A tile (128x32 halves = 8KB): 2 async b128 per thread ----
#pragma unroll
    for (int part = 0; part < 2; ++part) {
      int r  = (tid >> 2) + part * 64;
      int sg = (tid & 3) * 8;                       // halves
      async_ld_b128(lds_of(&As[r][sg]),
                    (unsigned)((((size_t)(m0 + r)) * H_ + kk * 32 + sg) * 2), A);
    }
    // ---- stage W tile (32x128 halves = 8KB): 2 async b128 per thread ----
#pragma unroll
    for (int part = 0; part < 2; ++part) {
      int r  = (tid >> 4) + part * 16;
      int sg = (tid & 15) * 8;
      async_ld_b128(lds_of(&Ws[r][sg]),
                    (unsigned)((((size_t)(kk * 32 + r)) * H_ + n0 + sg) * 2), W);
    }
    if (kk + 1 < H_ / 32)   // warm L2 for the next K slab
      __builtin_prefetch(A + ((size_t)(m0 + (tid >> 1)) * H_ + (kk + 1) * 32), 0, 0);
    wait_async0();
    __syncthreads();

    // A fragment for this wave's 16 rows
    v16h a;
    {
      int m = wid * 16 + (lane & 15);
      int kh = (lane >> 4) * 16;
#pragma unroll
      for (int j = 0; j < 8; ++j) a[j] = As[m][kh + j];
#pragma unroll
      for (int j = 0; j < 8; ++j) a[8 + j] = As[m][kh + 8 + j];
    }
#pragma unroll
    for (int nt = 0; nt < 8; ++nt) {
      v16h b;
      int n = nt * 16 + (lane & 15);
      int kb = (lane >> 4) * 16;
#pragma unroll
      for (int j = 0; j < 16; ++j) b[j] = Ws[kb + j][n];
      acc[nt] = wmma16x16x32(a, b, acc[nt]);
    }
    __syncthreads();
  }

  const int mrow = m0 + wid * 16 + (lane >> 4) * 8;
#pragma unroll
  for (int nt = 0; nt < 8; ++nt) {
    int n = n0 + nt * 16 + (lane & 15);
    float bs = bias[n];
#pragma unroll
    for (int r = 0; r < 8; ++r) {
      float v = acc[nt][r] * scale + bs;
      if (OUT_F32) Cf[(size_t)(mrow + r) * H_ + n] = v;
      else         Ch[(size_t)(mrow + r) * H_ + n] = (_Float16)v;
    }
  }
}

// ---------------------------------------------------------------------------
// Attention core: one wave per (batch, 16-query-row tile).
//   TDM stages q tile + double-buffered k tiles;  S = q@k^T (WMMA) -> LDS
//   masked softmax -> attn (f32) + P (f16 LDS)
//   TDM double-buffers v blocks;  X = P@v * (1/sqrt(TK)) (WMMA) -> f16 ws
// ---------------------------------------------------------------------------

__global__ __launch_bounds__(32)
void attn_kernel(const _Float16* __restrict__ Qh,
                 const _Float16* __restrict__ Kh,
                 const _Float16* __restrict__ Vh,
                 const unsigned char* __restrict__ mask,
                 float* __restrict__ attnOut,
                 _Float16* __restrict__ Xh) {
  __shared__ _Float16 QV[16][H_];       // 16KB: q tile (ph1) / v dbl-buf (ph3)
  __shared__ _Float16 Kt[2][16][H_];    // 32KB: k tile double buffer
  __shared__ float    Sc[16][TK_];      // 32KB: score strip
  __shared__ _Float16 Ps[16][TK_];      // 16KB: probabilities (f16)

  const int bb   = blockIdx.x >> 6;        // 64 q-tiles per batch
  const int qt   = blockIdx.x & 63;
  const int lane = threadIdx.x;

  const _Float16* qbase = Qh + ((size_t)bb * TQ_ + qt * 16) * H_;
  const _Float16* kbase = Kh + (size_t)bb * TK_ * H_;
  const _Float16* vbase = Vh + (size_t)bb * TK_ * H_;

#ifndef HAVE_TDM
  // fallback staging for q (no TDM available)
  for (int i = lane; i < 16 * H_; i += 32) QV[i / H_][i % H_] = qbase[i];
#endif

  // -------- phase 1: scores S[16 x 512] --------
#ifdef HAVE_TDM
  tdm_load_2d(lds_of(&QV[0][0]), qbase, H_, 16, H_, H_, B_ * TQ_);
  tdm_load_2d(lds_of(&Kt[0][0][0]), kbase, H_, 16, H_, H_, B_ * TK_);
#endif
  for (int nt = 0; nt < TK_ / 16; ++nt) {
#ifdef HAVE_TDM
    if (nt + 1 < TK_ / 16) {
      wait_ds0();                               // done reading Kt[(nt+1)&1]
      tdm_load_2d(lds_of(&Kt[(nt + 1) & 1][0][0]),
                  kbase + (size_t)(nt + 1) * 16 * H_, H_, 16, H_, H_, B_ * TK_);
      wait_tensor1();                           // q + k[nt] resident
    } else {
      wait_tensor0();
    }
#else
    for (int i = lane; i < 16 * H_; i += 32)
      Kt[nt & 1][i / H_][i % H_] = kbase[(size_t)nt * 16 * H_ + i];
#endif
    v8f acc = {};
#pragma unroll 4
    for (int ks = 0; ks < H_ / 32; ++ks) {
      int k0 = ks * 32;
      v16h a, b;
      {
        int m = lane & 15, kh = (lane >> 4) * 16;
#pragma unroll
        for (int j = 0; j < 8; ++j) a[j] = QV[m][k0 + kh + j];
#pragma unroll
        for (int j = 0; j < 8; ++j) a[8 + j] = QV[m][k0 + kh + 8 + j];
      }
      {
        const _Float16* p = &Kt[nt & 1][lane & 15][k0 + (lane >> 4) * 16];
#pragma unroll
        for (int j = 0; j < 16; ++j) b[j] = p[j];
      }
      acc = wmma16x16x32(a, b, acc);
    }
    int m = (lane >> 4) * 8;
    int n = nt * 16 + (lane & 15);
#pragma unroll
    for (int r = 0; r < 8; ++r) Sc[m + r][n] = acc[r];
  }
  __syncthreads();

  // -------- phase 2: masked softmax (row per lane, lanes 0..15) --------
  if (lane < 16) {
    const unsigned char* mrow = mask + (size_t)bb * TK_;
    float mx = -3.0e38f;
    for (int k = 0; k < TK_; ++k) {
      float s = mrow[k] ? -3.0e38f : Sc[lane][k];
      Sc[lane][k] = s;
      mx = fmaxf(mx, s);
    }
    float sum = 0.0f;
    for (int k = 0; k < TK_; ++k) {
      float e = mrow[k] ? 0.0f : __expf(Sc[lane][k] - mx);
      Sc[lane][k] = e;
      sum += e;
    }
    float inv = 1.0f / sum;
    float* arow = attnOut + ((size_t)bb * TQ_ + qt * 16 + lane) * TK_;
    for (int k = 0; k < TK_; ++k) {
      float p = Sc[lane][k] * inv;
      arow[k]     = p;                 // attn output (f32)
      Ps[lane][k] = (_Float16)p;       // operand for P @ v
    }
  }
  __syncthreads();

  // -------- phase 3: X = P @ v * XSCALE, 128-col chunks, v double-buffered ---
  _Float16 (*Vs)[32][128] = reinterpret_cast<_Float16 (*)[32][128]>(&QV[0][0]);
  _Float16* xbase = Xh + ((size_t)bb * TQ_ + qt * 16) * H_;

  for (int nc = 0; nc < H_ / 128; ++nc) {
#ifdef HAVE_TDM
    wait_ds0();
    tdm_load_2d(lds_of(&Vs[0][0][0]), vbase + (size_t)nc * 128,
                128, 32, H_, H_, B_ * TK_);
#endif
    v8f acc[8] = {};
#pragma unroll 1
    for (int ks = 0; ks < TK_ / 32; ++ks) {
#ifdef HAVE_TDM
      if (ks + 1 < TK_ / 32) {
        wait_ds0();
        tdm_load_2d(lds_of(&Vs[(ks + 1) & 1][0][0]),
                    vbase + (size_t)(ks + 1) * 32 * H_ + nc * 128,
                    128, 32, H_, H_, B_ * TK_);
        wait_tensor1();
      } else {
        wait_tensor0();
      }
#else
      for (int i = lane; i < 32 * 128; i += 32)
        Vs[ks & 1][i / 128][i % 128] =
            vbase[(size_t)(ks * 32 + i / 128) * H_ + nc * 128 + (i % 128)];
#endif
      int k0 = ks * 32;
      v16h a;
      {
        int m = lane & 15, kh = (lane >> 4) * 16;
#pragma unroll
        for (int j = 0; j < 8; ++j) a[j] = Ps[m][k0 + kh + j];
#pragma unroll
        for (int j = 0; j < 8; ++j) a[8 + j] = Ps[m][k0 + kh + 8 + j];
      }
#pragma unroll
      for (int nt = 0; nt < 8; ++nt) {
        v16h b;
        int n = nt * 16 + (lane & 15);
        int kb = (lane >> 4) * 16;
#pragma unroll
        for (int j = 0; j < 16; ++j) b[j] = Vs[ks & 1][kb + j][n];
        acc[nt] = wmma16x16x32(a, b, acc[nt]);
      }
    }
    int m = (lane >> 4) * 8;
#pragma unroll
    for (int nt = 0; nt < 8; ++nt) {
      int n = nc * 128 + nt * 16 + (lane & 15);
#pragma unroll
      for (int r = 0; r < 8; ++r)
        xbase[(size_t)(m + r) * H_ + n] = (_Float16)(acc[nt][r] * XSCALE);
    }
  }
}

// ---------------------------------------------------------------------------
// host: orchestration
// ---------------------------------------------------------------------------

extern "C" void kernel_launch(void* const* d_in, const int* in_sizes, int n_in,
                              void* d_out, int out_size, void* d_ws,
                              size_t ws_size, hipStream_t stream) {
  const float* query  = (const float*)d_in[0];
  const float* keys   = (const float*)d_in[1];
  const float* values = (const float*)d_in[2];
  const int*   tpos   = (const int*)d_in[3];
  const int*   fpos   = (const int*)d_in[4];
  const unsigned char* mask = (const unsigned char*)d_in[5];
  const float* Wq = (const float*)d_in[6];
  const float* bq = (const float*)d_in[7];
  const float* Wk = (const float*)d_in[8];
  const float* bk = (const float*)d_in[9];
  const float* Wv = (const float*)d_in[10];
  const float* bv = (const float*)d_in[11];
  const float* Wo = (const float*)d_in[12];
  const float* bo = (const float*)d_in[13];

  float* out     = (float*)d_out;
  float* attnOut = out + (size_t)B_ * TQ_ * H_;   // [B,TQ,TK] after [B,TQ,512]

  _Float16* ws = (_Float16*)d_ws;
  size_t off = 0;
  const size_t WSZ = (size_t)H_ * H_;
  const size_t QSZ = (size_t)B_ * TQ_ * H_;
  const size_t KSZ = (size_t)B_ * TK_ * H_;
  _Float16* WqH  = ws + off; off += WSZ;
  _Float16* WkH  = ws + off; off += WSZ;
  _Float16* WvH  = ws + off; off += WSZ;
  _Float16* WoH  = ws + off; off += WSZ;
  _Float16* QinH = ws + off; off += QSZ;
  _Float16* KinH = ws + off; off += KSZ;
  _Float16* VinH = ws + off; off += KSZ;
  _Float16* Qh   = ws + off; off += QSZ;
  _Float16* Kh   = ws + off; off += KSZ;
  _Float16* Vh   = ws + off; off += KSZ;
  _Float16* Xh   = ws + off; off += QSZ;

  // ---- 1. prep ----
  cvt_f16_kernel<<<(int)(WSZ / 256), 256, 0, stream>>>(Wq, WqH, (int)WSZ);
  cvt_f16_kernel<<<(int)(WSZ / 256), 256, 0, stream>>>(Wk, WkH, (int)WSZ);
  cvt_f16_kernel<<<(int)(WSZ / 256), 256, 0, stream>>>(Wv, WvH, (int)WSZ);
  cvt_f16_kernel<<<(int)(WSZ / 256), 256, 0, stream>>>(Wo, WoH, (int)WSZ);
  prep_pe_kernel<<<(int)(QSZ / 256), 256, 0, stream>>>(query, fpos, QinH, TQ_, 1.0f);
  prep_pe_kernel<<<(int)(KSZ / 256), 256, 0, stream>>>(keys,  tpos, KinH, TK_, 1.385f);
  cvt_f16_kernel<<<(int)(KSZ / 256), 256, 0, stream>>>(values, VinH, (int)KSZ);

  // ---- 2. projections ----
  dim3 blk(256);
  gemm_f16_kernel<false><<<dim3(B_ * TQ_ / 128, H_ / 128), blk, 0, stream>>>(
      QinH, WqH, bq, nullptr, Qh, 1.0f);
  gemm_f16_kernel<false><<<dim3(B_ * TK_ / 128, H_ / 128), blk, 0, stream>>>(
      KinH, WkH, bk, nullptr, Kh, 1.0f);
  gemm_f16_kernel<false><<<dim3(B_ * TK_ / 128, H_ / 128), blk, 0, stream>>>(
      VinH, WvH, bv, nullptr, Vh, 1.0f);

  // ---- 3. attention core ----
  attn_kernel<<<dim3(B_ * (TQ_ / 16)), dim3(32), 0, stream>>>(
      Qh, Kh, Vh, mask, attnOut, Xh);

  // ---- 4. out = Xh @ Wo + bo ----
  gemm_f16_kernel<true><<<dim3(B_ * TQ_ / 128, H_ / 128), blk, 0, stream>>>(
      Xh, WoH, bo, out, nullptr, 1.0f);
}
